// GRU_32512902431283
// MI455X (gfx1250) — matrix operand order repaired and checked by reading, not compiled, not run on hardware
//
#include <hip/hip_runtime.h>
#include <stdint.h>

// ---------------------------------------------------------------------------
// MI455X (gfx1250) fused 64-layer GRU.
//  * wave32: one wave owns a 16-row batch tile across all 64 layers.
//  * v_wmma_f32_16x16x32_bf16 for all H=64 -> 3H=192 GEMMs (C starts at
//    inline 0, biases folded into the gate epilogue).
//  * h0 == 0 per layer  => no W_hh GEMM at t=0.
//  * reference reshape  => only batch rows [8192, 16384) reach the output.
//  * Layer 0 (F==1 rank-1 input GEMM) is peeled out of the hot loop via a
//    template parameter: the 63-iteration steady-state loop is pure WMMA.
//  * Fast path: weights pre-packed to bf16 records in d_ws, streamed per layer
//    with double-buffered global_load_async_to_lds_b128 (ASYNCcnt), hiding the
//    weight-stage latency behind the previous layer's WMMA work.
// ---------------------------------------------------------------------------

#define HH        64
#define NLAYERS   64
#define NG        192          // 3*H
#define TPB       128          // 4 waves
#define WAVES_WG  4
#define ROW0      8192         // first live batch row

// Per-layer packed weight record in d_ws / LDS buffer:
//   [0     , 24576)  Wi bf16 [192][64]
//   [24576 , 49152)  Wh bf16 [192][64]
//   [49152 , 49920)  bi f32  [192]
//   [49920 , 50688)  bh f32  [192]
#define RECO_WI   0
#define RECO_WH   24576
#define RECO_BI   49152
#define RECO_BH   49920
#define REC       50688
#define WS_REQ    ((size_t)NLAYERS * REC)

#define LDS_SYNC_TOTAL   (REC + WAVES_WG * 4096)        // single buffer + seq
#define LDS_ASYNC_TOTAL  (2 * REC + WAVES_WG * 4096)    // double buffer + seq

typedef __attribute__((ext_vector_type(16))) __bf16 v16bf;
typedef __attribute__((ext_vector_type(8)))  float  v8f;

union Frag32B { uint4 q[2]; v16bf v; };

__device__ __forceinline__ uint32_t pk_bf16(float a, float b) {
  uint32_t ua = __builtin_bit_cast(uint32_t, a);
  uint32_t ub = __builtin_bit_cast(uint32_t, b);
  uint32_t ra = (ua + 0x7FFFu + ((ua >> 16) & 1u)) >> 16;   // RNE
  uint32_t rb = (ub + 0x7FFFu + ((ub >> 16) & 1u)) >> 16;
  return (rb << 16) | (ra & 0xFFFFu);
}
__device__ __forceinline__ uint16_t to_bf16(float a) {
  uint32_t u = __builtin_bit_cast(uint32_t, a);
  return (uint16_t)((u + 0x7FFFu + ((u >> 16) & 1u)) >> 16);
}
__device__ __forceinline__ float from_bf16(uint16_t h) {
  return __builtin_bit_cast(float, (uint32_t)h << 16);
}
__device__ __forceinline__ float sigm(float x) { return 1.f / (1.f + __expf(-x)); }
__device__ __forceinline__ float tanh_f(float x) { return 2.f / (1.f + __expf(-2.f * x)) - 1.f; }

// A operand (16x32 bf16 chunk): lane L holds row M=L%16;
// k = (i>=8?16:0) + (L>=16?8:0) + (i&7)  ->  two ds_load_b128 per lane.
__device__ __forceinline__ v16bf load_a_frag(const uint8_t* base, int lane, int chunk) {
  const int row = lane & 15;
  const int kb  = chunk * 32 + ((lane >> 4) << 3);
  const uint8_t* p = base + row * 128 + kb * 2;
  Frag32B f;
  f.q[0] = *(const uint4*)(p);
  f.q[1] = *(const uint4*)(p + 32);   // +16 k-elements
  return f.v;
}

// B operand (32x16 bf16): lane L holds col N=L%16,
// k = (i>=8?16:0) + (L>=16?8:0) + (i&7) of row-major W[192][64] -> two ds_load_b128 per lane.
__device__ __forceinline__ v16bf load_b_frag(const uint8_t* w, int lane, int ntile, int chunk) {
  const int wrow = ntile * 16 + (lane & 15);
  const int kb   = chunk * 32 + ((lane >> 4) << 3);
  const uint8_t* p = w + wrow * 128 + kb * 2;
  Frag32B f;
  f.q[0] = *(const uint4*)(p);
  f.q[1] = *(const uint4*)(p + 32);
  return f.v;
}

__device__ __forceinline__ v8f wmma_bf16(v16bf a, v16bf b, v8f c) {
  return __builtin_amdgcn_wmma_f32_16x16x32_bf16(
      false, a, false, b, (short)0, c, false, false);
}

// One GRU layer for this wave's 16-row tile. seq0/seq1 hold the incoming
// h(t=0)/h(t=1) of the previous layer (bf16 16x64 row-major) and are
// overwritten in place with this layer's outputs. FIRST (layer 0) replaces
// the input GEMM with the F==1 rank-1 product.
template <bool FIRST>
__device__ __forceinline__ void gru_layer_compute(
    const uint8_t* wiL, const uint8_t* whL,
    const float* biL, const float* bhL,
    uint8_t* seq0, uint8_t* seq1,
    const float* __restrict__ W_ih0,
    const float* xv0, const float* xv1,
    int lane, int lo, int n16)
{
  // ---------------- t = 0  (h_prev == 0: no Wh GEMM) ----------------
  v16bf as0, as1;
  if (!FIRST) { as0 = load_a_frag(seq0, lane, 0); as1 = load_a_frag(seq0, lane, 1); }

  for (int ct = 0; ct < 4; ++ct) {
    const int g = ct * 16 + n16;
    v8f accR = {}, accZ = {}, accN = {};           // C starts at inline 0
    if (FIRST) {                                   // rank-1 input GEMM (F==1)
      const float wr = W_ih0[g], wz = W_ih0[64 + g], wn = W_ih0[128 + g];
#pragma unroll
      for (int e = 0; e < 8; ++e) {
        accR[e] = xv0[e] * wr; accZ[e] = xv0[e] * wz; accN[e] = xv0[e] * wn;
      }
    } else {
      accR = wmma_bf16(as0, load_b_frag(wiL, lane, ct,     0), accR);
      accR = wmma_bf16(as1, load_b_frag(wiL, lane, ct,     1), accR);
      accZ = wmma_bf16(as0, load_b_frag(wiL, lane, 4 + ct, 0), accZ);
      accZ = wmma_bf16(as1, load_b_frag(wiL, lane, 4 + ct, 1), accZ);
      accN = wmma_bf16(as0, load_b_frag(wiL, lane, 8 + ct, 0), accN);
      accN = wmma_bf16(as1, load_b_frag(wiL, lane, 8 + ct, 1), accN);
    }
    const float br  = biL[g] + bhL[g];
    const float bz  = biL[64 + g] + bhL[64 + g];
    const float bni = biL[128 + g];
    const float bnh = bhL[128 + g];                // h_n == bh_n at t=0
#pragma unroll
    for (int e = 0; e < 8; ++e) {
      const float rg = sigm(accR[e] + br);
      const float zg = sigm(accZ[e] + bz);
      const float ng = tanh_f(accN[e] + bni + rg * bnh);
      *(uint16_t*)(seq0 + (e + 8 * lo) * 128 + g * 2) = to_bf16((1.f - zg) * ng);
    }
  }

  // ---------------- t = 1  (full step) ----------------
  v16bf ah0 = load_a_frag(seq0, lane, 0);          // h1
  v16bf ah1 = load_a_frag(seq0, lane, 1);
  v16bf bs0, bs1;
  if (!FIRST) { bs0 = load_a_frag(seq1, lane, 0); bs1 = load_a_frag(seq1, lane, 1); }

  for (int ct = 0; ct < 4; ++ct) {
    const int g = ct * 16 + n16;
    v8f accR = {}, accZ = {}, accIN = {}, accHN = {};
    if (FIRST) {
      const float wr = W_ih0[g], wz = W_ih0[64 + g], wn = W_ih0[128 + g];
#pragma unroll
      for (int e = 0; e < 8; ++e) {
        accR[e] = xv1[e] * wr; accZ[e] = xv1[e] * wz; accIN[e] = xv1[e] * wn;
      }
    } else {
      accR  = wmma_bf16(bs0, load_b_frag(wiL, lane, ct,     0), accR);
      accR  = wmma_bf16(bs1, load_b_frag(wiL, lane, ct,     1), accR);
      accZ  = wmma_bf16(bs0, load_b_frag(wiL, lane, 4 + ct, 0), accZ);
      accZ  = wmma_bf16(bs1, load_b_frag(wiL, lane, 4 + ct, 1), accZ);
      accIN = wmma_bf16(bs0, load_b_frag(wiL, lane, 8 + ct, 0), accIN);
      accIN = wmma_bf16(bs1, load_b_frag(wiL, lane, 8 + ct, 1), accIN);
    }
    accR  = wmma_bf16(ah0, load_b_frag(whL, lane, ct,     0), accR);
    accR  = wmma_bf16(ah1, load_b_frag(whL, lane, ct,     1), accR);
    accZ  = wmma_bf16(ah0, load_b_frag(whL, lane, 4 + ct, 0), accZ);
    accZ  = wmma_bf16(ah1, load_b_frag(whL, lane, 4 + ct, 1), accZ);
    accHN = wmma_bf16(ah0, load_b_frag(whL, lane, 8 + ct, 0), accHN);
    accHN = wmma_bf16(ah1, load_b_frag(whL, lane, 8 + ct, 1), accHN);

    const float br  = biL[g] + bhL[g];
    const float bz  = biL[64 + g] + bhL[64 + g];
    const float bni = biL[128 + g];
    const float bnh = bhL[128 + g];
#pragma unroll
    for (int e = 0; e < 8; ++e) {
      const float rg = sigm(accR[e] + br);
      const float zg = sigm(accZ[e] + bz);
      const float ng = tanh_f(accIN[e] + bni + rg * (accHN[e] + bnh));
      const float hp = from_bf16(*(const uint16_t*)(seq0 + (e + 8 * lo) * 128 + g * 2));
      *(uint16_t*)(seq1 + (e + 8 * lo) * 128 + g * 2) =
          to_bf16((1.f - zg) * ng + zg * hp);
    }
  }
}

__device__ __forceinline__ void final_linear(
    const uint8_t* seq0, const uint8_t* seq1,
    const float* __restrict__ Wf, float bf0,
    int lo, int n16, int R, float* __restrict__ out)
{
  const uint8_t* hrow = (lo == 0 ? seq0 : seq1) + n16 * 128;
  float acc = bf0;
#pragma unroll
  for (int k = 0; k < HH; k += 2) {
    const uint32_t p2 = *(const uint32_t*)(hrow + k * 2);
    acc += from_bf16((uint16_t)p2)         * Wf[k];
    acc += from_bf16((uint16_t)(p2 >> 16)) * Wf[k + 1];
  }
  const int row = R + n16;
  out[2 * (row - ROW0) + lo] = acc;
}

// ---------------------------------------------------------------------------
// Pre-pass: pack per-layer bf16 weight records into d_ws.
// ---------------------------------------------------------------------------
__global__ void convert_weights_kernel(
    const float* __restrict__ W_hh0,
    const float* __restrict__ b_ih0, const float* __restrict__ b_hh0,
    const float* __restrict__ W_ih,  const float* __restrict__ W_hh,
    const float* __restrict__ b_ih,  const float* __restrict__ b_hh,
    uint8_t* __restrict__ ws)
{
  const int nt = blockDim.x * gridDim.x;
  const int t0 = blockIdx.x * blockDim.x + threadIdx.x;
  const int WPM = NG * HH / 2;                       // u32 words per matrix
  for (int i = t0; i < NLAYERS * WPM; i += nt) {
    const int l = i / WPM, w = i % WPM;
    uint8_t* rec = ws + (size_t)l * REC;
    uint32_t wiv = 0u;
    if (l > 0) {
      const float* s = W_ih + (size_t)(l - 1) * NG * HH + w * 2;
      wiv = pk_bf16(s[0], s[1]);
    }
    ((uint32_t*)(rec + RECO_WI))[w] = wiv;
    const float* sh = (l == 0) ? (W_hh0 + w * 2)
                               : (W_hh + (size_t)(l - 1) * NG * HH + w * 2);
    ((uint32_t*)(rec + RECO_WH))[w] = pk_bf16(sh[0], sh[1]);
  }
  for (int i = t0; i < NLAYERS * NG; i += nt) {
    const int l = i / NG, j = i % NG;
    uint8_t* rec = ws + (size_t)l * REC;
    ((float*)(rec + RECO_BI))[j] = (l == 0) ? b_ih0[j] : b_ih[(l - 1) * NG + j];
    ((float*)(rec + RECO_BH))[j] = (l == 0) ? b_hh0[j] : b_hh[(l - 1) * NG + j];
  }
}

// ---------------------------------------------------------------------------
// Fast path: double-buffered async weight staging (global_load_async_to_lds).
// ---------------------------------------------------------------------------
__global__ __launch_bounds__(TPB) void gru_fused_async_kernel(
    const float* __restrict__ x,     const float* __restrict__ W_ih0,
    const float* __restrict__ Wf,    const float* __restrict__ bfin,
    const uint8_t* __restrict__ wrec,
    float* __restrict__ out)
{
  extern __shared__ uint8_t smem[];
  uint8_t* buf0 = smem;
  uint8_t* buf1 = smem + REC;

  const int tid  = threadIdx.x;
  const int lane = tid & 31;
  const int wave = tid >> 5;
  const int lo   = lane >> 4;
  const int n16  = lane & 15;

  uint8_t* seq0 = smem + 2 * REC + wave * 4096;
  uint8_t* seq1 = seq0 + 2048;

  const int R = ROW0 + (blockIdx.x * WAVES_WG + wave) * 16;

  float xv0[8], xv1[8];
#pragma unroll
  for (int r = 0; r < 8; ++r) {
    const int m = R + r + 8 * lo;
    xv0[r] = x[m * 2 + 0];
    xv1[r] = x[m * 2 + 1];
  }

  // Issue one layer record copy (50688 B) as per-lane 16B async transfers.
  auto issue_copy = [&](int l) {
    uint8_t* dst = (l & 1) ? buf1 : buf0;
    const uint8_t* src = wrec + (size_t)l * REC;
    for (int off = tid * 16; off < REC; off += TPB * 16) {
      const uint32_t lds32 = (uint32_t)(uintptr_t)(dst + off);   // LDS byte addr
      const uint64_t g64   = (uint64_t)(uintptr_t)(src + off);
      asm volatile("global_load_async_to_lds_b128 %0, %1, off"
                   :: "v"(lds32), "v"(g64) : "memory");
    }
  };

  issue_copy(0);
  asm volatile("s_wait_asynccnt 0x0" ::: "memory");
  __syncthreads();

  // ---- layer 0 peeled (rank-1 input path) ----
  issue_copy(1);
  gru_layer_compute<true>(buf0 + RECO_WI, buf0 + RECO_WH,
                          (const float*)(buf0 + RECO_BI),
                          (const float*)(buf0 + RECO_BH),
                          seq0, seq1, W_ih0, xv0, xv1, lane, lo, n16);
  asm volatile("s_wait_asynccnt 0x0" ::: "memory");
  __syncthreads();

  // ---- steady state: pure WMMA loop ----
  for (int l = 1; l < NLAYERS; ++l) {
    if (l + 1 < NLAYERS) issue_copy(l + 1);        // overlap with compute

    const uint8_t* cur = (l & 1) ? buf1 : buf0;
    gru_layer_compute<false>(cur + RECO_WI, cur + RECO_WH,
                             (const float*)(cur + RECO_BI),
                             (const float*)(cur + RECO_BH),
                             seq0, seq1, W_ih0, xv0, xv1, lane, lo, n16);

    asm volatile("s_wait_asynccnt 0x0" ::: "memory");
    __syncthreads();
  }

  final_linear(seq0, seq1, Wf, bfin[0], lo, n16, R, out);
}

// ---------------------------------------------------------------------------
// Fallback path (ws too small): per-layer fp32 load + convert into LDS.
// ---------------------------------------------------------------------------
__global__ __launch_bounds__(TPB) void gru_fused_sync_kernel(
    const float* __restrict__ x,
    const float* __restrict__ W_ih0, const float* __restrict__ W_hh0,
    const float* __restrict__ b_ih0, const float* __restrict__ b_hh0,
    const float* __restrict__ W_ih,  const float* __restrict__ W_hh,
    const float* __restrict__ b_ih,  const float* __restrict__ b_hh,
    const float* __restrict__ Wf,    const float* __restrict__ bfin,
    float* __restrict__ out)
{
  extern __shared__ uint8_t smem[];
  uint8_t* wiL = smem + RECO_WI;
  uint8_t* whL = smem + RECO_WH;
  float*   biL = (float*)(smem + RECO_BI);
  float*   bhL = (float*)(smem + RECO_BH);

  const int tid  = threadIdx.x;
  const int lane = tid & 31;
  const int wave = tid >> 5;
  const int lo   = lane >> 4;
  const int n16  = lane & 15;

  uint8_t* seq0 = smem + REC + wave * 4096;
  uint8_t* seq1 = seq0 + 2048;

  const int R = ROW0 + (blockIdx.x * WAVES_WG + wave) * 16;

  float xv0[8], xv1[8];
#pragma unroll
  for (int r = 0; r < 8; ++r) {
    const int m = R + r + 8 * lo;
    xv0[r] = x[m * 2 + 0];
    xv1[r] = x[m * 2 + 1];
  }

  auto load_weights = [&](int l) {
    const float* Wh_g = (l == 0) ? W_hh0 : (W_hh + (size_t)(l - 1) * NG * HH);
    const float* Wi_g = (l == 0) ? (const float*)0 : (W_ih + (size_t)(l - 1) * NG * HH);
    const float* bi_g = (l == 0) ? b_ih0 : (b_ih + (size_t)(l - 1) * NG);
    const float* bh_g = (l == 0) ? b_hh0 : (b_hh + (size_t)(l - 1) * NG);
    for (int idx = tid; idx < NG * HH / 4; idx += TPB) {
      float4 v = ((const float4*)Wh_g)[idx];
      ((uint2*)whL)[idx] = make_uint2(pk_bf16(v.x, v.y), pk_bf16(v.z, v.w));
      if (Wi_g) {
        float4 w = ((const float4*)Wi_g)[idx];
        ((uint2*)wiL)[idx] = make_uint2(pk_bf16(w.x, w.y), pk_bf16(w.z, w.w));
      }
    }
    if (tid < NG) { biL[tid] = bi_g[tid]; bhL[tid] = bh_g[tid]; }
    if (l + 1 < NLAYERS) {
      __builtin_prefetch(W_hh + (size_t)l * NG * HH + tid * 96, 0, 1);
      __builtin_prefetch(W_ih + (size_t)l * NG * HH + tid * 96, 0, 1);
    }
  };

  // ---- layer 0 peeled ----
  load_weights(0);
  __syncthreads();
  gru_layer_compute<true>(wiL, whL, biL, bhL, seq0, seq1,
                          W_ih0, xv0, xv1, lane, lo, n16);

  // ---- steady state ----
  for (int l = 1; l < NLAYERS; ++l) {
    __syncthreads();                 // previous layer done reading weights
    load_weights(l);
    __syncthreads();
    gru_layer_compute<false>(wiL, whL, biL, bhL, seq0, seq1,
                             W_ih0, xv0, xv1, lane, lo, n16);
  }

  final_linear(seq0, seq1, Wf, bfin[0], lo, n16, R, out);
}

extern "C" void kernel_launch(void* const* d_in, const int* in_sizes, int n_in,
                              void* d_out, int out_size, void* d_ws, size_t ws_size,
                              hipStream_t stream) {
  const float* x     = (const float*)d_in[0];
  const float* W_ih0 = (const float*)d_in[1];
  const float* W_hh0 = (const float*)d_in[2];
  const float* b_ih0 = (const float*)d_in[3];
  const float* b_hh0 = (const float*)d_in[4];
  const float* W_ih  = (const float*)d_in[5];
  const float* W_hh  = (const float*)d_in[6];
  const float* b_ih  = (const float*)d_in[7];
  const float* b_hh  = (const float*)d_in[8];
  const float* Wf    = (const float*)d_in[9];
  const float* bfin  = (const float*)d_in[10];

  (void)hipFuncSetAttribute((const void*)gru_fused_async_kernel,
                            hipFuncAttributeMaxDynamicSharedMemorySize,
                            LDS_ASYNC_TOTAL);
  (void)hipFuncSetAttribute((const void*)gru_fused_sync_kernel,
                            hipFuncAttributeMaxDynamicSharedMemorySize,
                            LDS_SYNC_TOTAL);

  const int rows = 8192;                        // only live batch rows
  const int grid = rows / (WAVES_WG * 16);      // 128 workgroups

  if (ws_size >= WS_REQ && d_ws != nullptr) {
    convert_weights_kernel<<<192, 256, 0, stream>>>(
        W_hh0, b_ih0, b_hh0, W_ih, W_hh, b_ih, b_hh, (uint8_t*)d_ws);
    gru_fused_async_kernel<<<grid, TPB, LDS_ASYNC_TOTAL, stream>>>(
        x, W_ih0, Wf, bfin, (const uint8_t*)d_ws, (float*)d_out);
  } else {
    gru_fused_sync_kernel<<<grid, TPB, LDS_SYNC_TOTAL, stream>>>(
        x, W_ih0, W_hh0, b_ih0, b_hh0, W_ih, W_hh, b_ih, b_hh, Wf, bfin,
        (float*)d_out);
  }
}
